// Money_former_MLA_DINT_cog_attn_MTP_28475633172569
// MI455X (gfx1250) — matrix operand order repaired and checked by reading, not compiled, not run on hardware
//
#include <hip/hip_runtime.h>

// ---------------------------------------------------------------------------
// MLA + differential sign-softmax attention for MI455X (gfx1250, wave32).
// All matrix ops use v_wmma_f32_16x16x32_bf16; operands arranged so each
// lane's fragment is a contiguous global load (no transpose loads needed).
// ---------------------------------------------------------------------------

typedef __attribute__((ext_vector_type(16))) __bf16 bf16x16;
typedef __attribute__((ext_vector_type(8)))  __bf16 bf16x8;
typedef __attribute__((ext_vector_type(8)))  float  f32x8;

constexpr int   cB   = 2;
constexpr int   cL   = 1024;   // NS*SP
constexpr int   cD   = 1024;
constexpr int   cH   = 16;
constexpr int   cHD  = 64;
constexpr int   cKVC = 256;
constexpr int   cQC  = 384;
constexpr int   cRD  = 32;
constexpr int   cSP  = 256;
constexpr float cEPS = 1.1920929e-7f;          // float32 machine eps
constexpr float cSCALE = 0.14433756729740643f; // ((64+32)/2)^-0.5
constexpr float cLAMBDA_INIT = 0.2f;

// ---------------------------------------------------------------------------
// Generic batched WMMA GEMM:  C[M,N] = alpha * A[M,K] x B[K,N]
// A   : bf16 row-major (M x K), K % 32 == 0, rows 16B aligned
// Bt  : bf16 row-major (N x K)  (i.e. B pre-transposed), rows 32B aligned
// C   : fp32 row-major (M x N), N % 32 == 0, M % 64 == 0 per grid sizing
// One wave computes a 16x32 tile (A-frag reused across two B-frags).
// block = (32,4) -> 4 waves;  grid = (N/32, M/64, batch)
// ---------------------------------------------------------------------------
__global__ __launch_bounds__(128)
void wmma_gemm_bf16(const __bf16* __restrict__ A, const __bf16* __restrict__ Bt,
                    float* __restrict__ C, int M, int N, int K, float alpha,
                    long sA, long sB, long sC)
{
  const long batch = blockIdx.z;
  A  += batch * sA;
  Bt += batch * sB;
  C  += batch * sC;

  const int lane = threadIdx.x;           // 0..31
  const int wave = threadIdx.y;           // 0..3
  const int m0 = (blockIdx.y * 4 + wave) * 16;
  const int n0 = blockIdx.x * 32;
  if (m0 >= M) return;

  const int mn = lane & 15;               // M index (A frag) / N index (B frag)
  const int hi = lane >> 4;               // lane-group select

  const __bf16* arow  = A  + (long)(m0 + mn) * K;        // row of A
  const __bf16* brow0 = Bt + (long)(n0 + mn) * K;        // column n0+mn of B
  const __bf16* brow1 = Bt + (long)(n0 + 16 + mn) * K;   // column n0+16+mn

  f32x8 acc0 = {};
  f32x8 acc1 = {};

  for (int k0 = 0; k0 < K; k0 += 32) {
    union { bf16x16 v; bf16x8 h[2]; } af, b0f, b1f;
    // A frag: lanes 0-15 hold K = k0+[0..15]; lanes 16-31 K = k0+[16..31]
    af.h[0] = *(const bf16x8*)(arow + k0 + 16 * hi);
    af.h[1] = *(const bf16x8*)(arow + k0 + 16 * hi + 8);
    // B frag: lanes 0-15 hold K = k0+[0..15]; lanes 16-31 K = k0+[16..31]
    b0f.v = *(const bf16x16*)(brow0 + k0 + 16 * hi);
    b1f.v = *(const bf16x16*)(brow1 + k0 + 16 * hi);

    if (k0 + 32 < K) __builtin_prefetch(arow + k0 + 32, 0, 1);

    acc0 = __builtin_amdgcn_wmma_f32_16x16x32_bf16(false, af.v, false, b0f.v,
                                                   (short)0, acc0, false, false);
    acc1 = __builtin_amdgcn_wmma_f32_16x16x32_bf16(false, af.v, false, b1f.v,
                                                   (short)0, acc1, false, false);
  }

  // C/D layout: VGPR r, lanes 0-15 -> M=r ; lanes 16-31 -> M=8+r ; N=lane&15
  float* crow = C + (long)m0 * N + n0;
  for (int r = 0; r < 8; ++r) {
    const int ml = r + 8 * hi;
    crow[(long)ml * N + mn]      = alpha * acc0[r];
    crow[(long)ml * N + 16 + mn] = alpha * acc1[r];
  }
}

// ---------------------------------------------------------------------------
// W (K x N, fp32, row-major) -> Wt (N x K, bf16, row-major)
// ---------------------------------------------------------------------------
__global__ void transpose_to_bf16(const float* __restrict__ W,
                                  __bf16* __restrict__ Wt, int K, int N)
{
  long idx = (long)blockIdx.x * 256 + threadIdx.x;
  if (idx >= (long)K * N) return;
  int n = (int)(idx / K);
  int k = (int)(idx % K);
  Wt[idx] = (__bf16)W[(long)k * N + n];
}

__global__ void to_bf16_kernel(const float* __restrict__ X,
                               __bf16* __restrict__ Y, long n)
{
  long idx = (long)blockIdx.x * 256 + threadIdx.x;
  if (idx < n) Y[idx] = (__bf16)X[idx];
}

// ---------------------------------------------------------------------------
// Row-wise RMS norm: Y[t, 0..ncols) = X[t*stride + c] * rsqrt(mean(x^2)+eps) * g[c]
// one block (256 threads) per row
// ---------------------------------------------------------------------------
__global__ void rms_rows(const float* __restrict__ X, const float* __restrict__ g,
                         __bf16* __restrict__ Y, int ncols, int stride, float eps)
{
  const int t = blockIdx.x;
  const float* row = X + (long)t * stride;
  __shared__ float red[256];
  float s = 0.f;
  for (int c = threadIdx.x; c < ncols; c += 256) { float v = row[c]; s += v * v; }
  red[threadIdx.x] = s;
  __syncthreads();
  for (int off = 128; off > 0; off >>= 1) {
    if (threadIdx.x < off) red[threadIdx.x] += red[threadIdx.x + off];
    __syncthreads();
  }
  const float scale = rsqrtf(red[0] / (float)ncols + eps);
  for (int c = threadIdx.x; c < ncols; c += 256)
    Y[(long)t * ncols + c] = (__bf16)(row[c] * scale * g[c]);
}

// ---------------------------------------------------------------------------
// Build Q/K (B,2H,L,64 bf16; dims 48..63 zero-padded) with segment RoPE,
// and V transposed (B,H,HD,L bf16) for the att@v GEMM B-operand.
// One thread per (b, l, h2).
// ---------------------------------------------------------------------------
__global__ void build_qkv_kernel(const float* __restrict__ Cb,   // (T, 288)
                                 const float* __restrict__ KV,   // (T, 2048)
                                 const float* __restrict__ QF,   // (T, 1536)
                                 const float* __restrict__ fc,   // (255, 8)
                                 const float* __restrict__ fs,   // (255, 8)
                                 __bf16* __restrict__ Qb,
                                 __bf16* __restrict__ Kb,
                                 __bf16* __restrict__ Vt)
{
  const int idx = blockIdx.x * 256 + threadIdx.x;        // B*L*2H
  if (idx >= cB * cL * 2 * cH) return;
  const int h2   = idx & 31;
  const int t    = idx >> 5;            // global token
  const int l    = t & (cL - 1);
  const int b    = t >> 10;
  const int p    = l & (cSP - 1);       // position in segment
  const int half = h2 & 1;
  const int h    = h2 >> 1;

  float cs[8], sn[8];
  for (int j = 0; j < 8; ++j) { cs[j] = 1.f; sn[j] = 0.f; }
  if (p > 0)
    for (int j = 0; j < 8; ++j) { cs[j] = fc[(p - 1) * 8 + j]; sn[j] = fs[(p - 1) * 8 + j]; }

  // ---- Q : qf (B,L,H,96); q = qf[...,:64] as (2H,32); rope = qf[...,64:96] as (2H,16)
  const float* qsrc = QF + (long)t * (cH * (cHD + cRD)) + h * (cHD + cRD);
  __bf16* qo = Qb + ((((long)(b * 32 + h2)) << 10) + l) * 64;
  for (int d = 0; d < 32; ++d) qo[d] = (__bf16)qsrc[half * 32 + d];
  {
    float r[16];
    for (int j = 0; j < 16; ++j) r[j] = qsrc[cHD + half * 16 + j];
    if (p > 0)
      for (int j = 0; j < 8; ++j) {
        float xr = r[2 * j], xi = r[2 * j + 1];
        r[2 * j]     = xr * cs[j] - xi * sn[j];
        r[2 * j + 1] = xr * sn[j] + xi * cs[j];
      }
    for (int j = 0; j < 16; ++j) qo[32 + j] = (__bf16)r[j];
  }
  for (int d = 48; d < 64; ++d) qo[d] = (__bf16)0.0f;

  // ---- K : kv (B,L,H,128); k = kv[...,:64] as (2H,32); rope from c[...,256:288] as (2H,16)
  const float* ksrc = KV + (long)t * (2 * cH * cHD) + h * (2 * cHD);
  __bf16* ko = Kb + ((((long)(b * 32 + h2)) << 10) + l) * 64;
  for (int d = 0; d < 32; ++d) ko[d] = (__bf16)ksrc[half * 32 + d];
  {
    const float* rsrc = Cb + (long)t * (cKVC + cRD) + cKVC;
    float r[16];
    for (int j = 0; j < 16; ++j) r[j] = rsrc[half * 16 + j];
    if (p > 0)
      for (int j = 0; j < 8; ++j) {
        float xr = r[2 * j], xi = r[2 * j + 1];
        r[2 * j]     = xr * cs[j] - xi * sn[j];
        r[2 * j + 1] = xr * sn[j] + xi * cs[j];
      }
    for (int j = 0; j < 16; ++j) ko[32 + j] = (__bf16)r[j];
  }
  for (int d = 48; d < 64; ++d) ko[d] = (__bf16)0.0f;

  // ---- V transposed: Vt[b,h,d,l] = kv[t, h*128 + 64 + d]
  if (half == 0) {
    const float* vsrc = ksrc + cHD;
    for (int d = 0; d < 64; ++d)
      Vt[(((long)(b * cH + h) * 64 + d) << 10) + l] = (__bf16)vsrc[d];
  }
}

// ---------------------------------------------------------------------------
// In-place sign-softmax over each score row (L=1024). One block per row.
// p = sign(a) * softmax(where(mask, -1e9, |a|))
// ---------------------------------------------------------------------------
__global__ void sign_softmax(float* __restrict__ P, const int* __restrict__ mask)
{
  const long row = blockIdx.x;             // over B*2H*L
  const int  i   = (int)(row & (cL - 1));
  float* p = P + row * (long)cL;
  const int* mrow = mask + (long)i * cL;
  const int tid = threadIdx.x;

  float a[4], e[4];
  float mx = -1e30f;
  for (int k = 0; k < 4; ++k) {
    const int j = tid + k * 256;
    const float v = p[j];
    a[k] = v;
    const float m = mrow[j] ? -1.0e9f : fabsf(v);
    e[k] = m;
    mx = fmaxf(mx, m);
  }

  __shared__ float red[256];
  red[tid] = mx;
  __syncthreads();
  for (int off = 128; off > 0; off >>= 1) {
    if (tid < off) red[tid] = fmaxf(red[tid], red[tid + off]);
    __syncthreads();
  }
  mx = red[0];
  __syncthreads();

  float sum = 0.f;
  for (int k = 0; k < 4; ++k) { e[k] = __expf(e[k] - mx); sum += e[k]; }
  red[tid] = sum;
  __syncthreads();
  for (int off = 128; off > 0; off >>= 1) {
    if (tid < off) red[tid] += red[tid + off];
    __syncthreads();
  }
  const float inv = 1.f / red[0];

  for (int k = 0; k < 4; ++k) {
    const int j = tid + k * 256;
    const float sg = (a[k] > 0.f) ? 1.f : ((a[k] < 0.f) ? -1.f : 0.f);
    p[j] = sg * e[k] * inv;
  }
}

// lam = exp(sum lq1*lk1) - exp(sum lq2*lk2) + LAMBDA_INIT
__global__ void lambda_kernel(const float* __restrict__ q1, const float* __restrict__ k1,
                              const float* __restrict__ q2, const float* __restrict__ k2,
                              float* __restrict__ out)
{
  if (threadIdx.x == 0) {
    float s1 = 0.f, s2 = 0.f;
    for (int i = 0; i < cHD / 2; ++i) { s1 += q1[i] * k1[i]; s2 += q2[i] * k2[i]; }
    out[0] = __expf(s1) - __expf(s2) + cLAMBDA_INIT;
  }
}

// g[b,h,j] = mean_i p0[b,2h,i,j];  grid = (B*H, L/256), block 256
__global__ void col_mean(const float* __restrict__ P, float* __restrict__ G)
{
  const int bh = blockIdx.x;                  // b*16 + h
  const int j  = blockIdx.y * 256 + threadIdx.x;
  const int b = bh >> 4, h = bh & 15;
  const float* p0 = P + ((long)(b * 32 + 2 * h) << 20);   // * L * L
  float s = 0.f;
  for (int i = 0; i < cL; ++i) s += p0[((long)i << 10) + j];
  G[((long)bh << 10) + j] = s * (1.0f / (float)cL);
}

// att[b,h,i,j] = mask ? 0 : p0 - lam*p1 + lam*g[b,h,j]   ->  bf16
__global__ void build_att(const float* __restrict__ P, const float* __restrict__ G,
                          const float* __restrict__ lamp, const int* __restrict__ mask,
                          __bf16* __restrict__ Ab)
{
  const long idx = (long)blockIdx.x * 256 + threadIdx.x;  // over B*H*L*L
  const int  j   = (int)(idx & (cL - 1));
  const long t   = idx >> 10;
  const int  i   = (int)(t & (cL - 1));
  const int  bh  = (int)(t >> 10);
  const int  b = bh >> 4, h = bh & 15;

  float v = 0.f;
  if (!mask[((long)i << 10) + j]) {
    const float lam = lamp[0];
    const long b0 = (((long)(b * 32 + 2 * h) << 10) + i) << 10;
    v = P[b0 + j] - lam * P[b0 + ((long)1 << 20) + j] + lam * G[((long)bh << 10) + j];
  }
  Ab[idx] = (__bf16)v;
}

// Head-RMS (eps 1e-5) + relayout: Ob[b*L+l, h*64+d]. One 64-thread block per (b,h,l).
__global__ void head_rms(const float* __restrict__ O1, const float* __restrict__ g,
                         __bf16* __restrict__ Ob)
{
  const int blk = blockIdx.x;                 // (b*H + h)*L + l
  const int d   = threadIdx.x;
  const float v = O1[(long)blk * 64 + d];
  __shared__ float red[64];
  red[d] = v * v;
  __syncthreads();
  for (int off = 32; off > 0; off >>= 1) {
    if (d < off) red[d] += red[d + off];
    __syncthreads();
  }
  const float scale = rsqrtf(red[0] * (1.0f / 64.0f) + 1e-5f);
  const int bh = blk >> 10;
  const int l  = blk & (cL - 1);
  const int b = bh >> 4, h = bh & 15;
  const long t = (long)b * cL + l;
  Ob[t * (cH * cHD) + h * 64 + d] = (__bf16)(v * scale * g[d]);
}

// ---------------------------------------------------------------------------
static void launch_gemm(const __bf16* A, const __bf16* Bt, float* C,
                        int M, int N, int K, float alpha, int batch,
                        long sA, long sB, long sC, hipStream_t stream)
{
  dim3 blk(32, 4);
  dim3 grd(N / 32, M / 64, batch);
  wmma_gemm_bf16<<<grd, blk, 0, stream>>>(A, Bt, C, M, N, K, alpha, sA, sB, sC);
}

extern "C" void kernel_launch(void* const* d_in, const int* in_sizes, int n_in,
                              void* d_out, int out_size, void* d_ws, size_t ws_size,
                              hipStream_t stream)
{
  (void)in_sizes; (void)n_in; (void)out_size; (void)ws_size;

  const float* x    = (const float*)d_in[0];
  const int*   mask = (const int*)  d_in[1];
  const float* fc   = (const float*)d_in[2];
  const float* fs   = (const float*)d_in[3];
  const float* Wkvd = (const float*)d_in[4];
  const float* Wqd  = (const float*)d_in[5];
  const float* Wkvu = (const float*)d_in[6];
  const float* Wqu  = (const float*)d_in[7];
  const float* Wo   = (const float*)d_in[8];
  const float* gkv  = (const float*)d_in[9];
  const float* gq   = (const float*)d_in[10];
  const float* gh   = (const float*)d_in[11];
  const float* lq1  = (const float*)d_in[12];
  const float* lk1  = (const float*)d_in[13];
  const float* lq2  = (const float*)d_in[14];
  const float* lk2  = (const float*)d_in[15];
  float* out = (float*)d_out;

  const long T = (long)cB * cL;   // 2048 tokens

  char* w = (char*)d_ws;
  auto carve = [&](size_t bytes) -> char* {
    char* p = w;
    w += (bytes + 255) & ~(size_t)255;
    return p;
  };

  float*  S     = (float*) carve((size_t)cB * 32 * cL * cL * 4);   // scores / p (256 MB)
  __bf16* Ab    = (__bf16*)carve((size_t)cB * cH * cL * cL * 2);   // att bf16 (64 MB)
  __bf16* xb    = (__bf16*)carve((size_t)T * cD * 2);
  __bf16* Wkvdt = (__bf16*)carve((size_t)(cKVC + cRD) * cD * 2);
  __bf16* Wqdt  = (__bf16*)carve((size_t)cQC * cD * 2);
  __bf16* Wkvut = (__bf16*)carve((size_t)2 * cH * cHD * cKVC * 2);
  __bf16* Wqut  = (__bf16*)carve((size_t)cH * (cHD + cRD) * cQC * 2);
  __bf16* Wot   = (__bf16*)carve((size_t)cD * cD * 2);
  float*  Cb    = (float*) carve((size_t)T * (cKVC + cRD) * 4);    // c = x @ W_kv_down
  __bf16* ckvn  = (__bf16*)carve((size_t)T * cKVC * 2);
  float*  Qd    = (float*) carve((size_t)T * cQC * 4);
  __bf16* qdn   = (__bf16*)carve((size_t)T * cQC * 2);
  float*  KV    = (float*) carve((size_t)T * 2 * cH * cHD * 4);
  float*  QF    = (float*) carve((size_t)T * cH * (cHD + cRD) * 4);
  __bf16* Qb    = (__bf16*)carve((size_t)cB * 32 * cL * 64 * 2);   // K-padded to 64
  __bf16* Kb    = (__bf16*)carve((size_t)cB * 32 * cL * 64 * 2);
  __bf16* Vt    = (__bf16*)carve((size_t)cB * cH * 64 * cL * 2);
  float*  G     = (float*) carve((size_t)cB * cH * cL * 4);
  float*  lam   = (float*) carve(256);
  float*  O1    = (float*) carve((size_t)cB * cH * cL * 64 * 4);
  __bf16* Ob    = (__bf16*)carve((size_t)T * cH * cHD * 2);

  // ---- weight transposition to bf16 (N x K) + activation conversion
  transpose_to_bf16<<<(cD * (cKVC + cRD) + 255) / 256, 256, 0, stream>>>(Wkvd, Wkvdt, cD, cKVC + cRD);
  transpose_to_bf16<<<(cD * cQC + 255) / 256, 256, 0, stream>>>(Wqd, Wqdt, cD, cQC);
  transpose_to_bf16<<<(cKVC * 2 * cH * cHD + 255) / 256, 256, 0, stream>>>(Wkvu, Wkvut, cKVC, 2 * cH * cHD);
  transpose_to_bf16<<<(cQC * cH * (cHD + cRD) + 255) / 256, 256, 0, stream>>>(Wqu, Wqut, cQC, cH * (cHD + cRD));
  transpose_to_bf16<<<(cD * cD + 255) / 256, 256, 0, stream>>>(Wo, Wot, cD, cD);
  to_bf16_kernel<<<(int)((T * cD + 255) / 256), 256, 0, stream>>>(x, xb, T * cD);

  // ---- down projections
  launch_gemm(xb, Wkvdt, Cb, (int)T, cKVC + cRD, cD, 1.f, 1, 0, 0, 0, stream);
  launch_gemm(xb, Wqdt,  Qd, (int)T, cQC,        cD, 1.f, 1, 0, 0, 0, stream);

  // ---- RMS norms
  rms_rows<<<(int)T, 256, 0, stream>>>(Cb, gkv, ckvn, cKVC, cKVC + cRD, cEPS);
  rms_rows<<<(int)T, 256, 0, stream>>>(Qd, gq, qdn, cQC, cQC, cEPS);

  // ---- up projections
  launch_gemm(ckvn, Wkvut, KV, (int)T, 2 * cH * cHD,     cKVC, 1.f, 1, 0, 0, 0, stream);
  launch_gemm(qdn,  Wqut,  QF, (int)T, cH * (cHD + cRD), cQC,  1.f, 1, 0, 0, 0, stream);

  // ---- q/k/v assembly with RoPE (and zero pad 48->64), v transposed
  build_qkv_kernel<<<(cB * cL * 2 * cH) / 256, 256, 0, stream>>>(Cb, KV, QF, fc, fs, Qb, Kb, Vt);

  // ---- scores: S[b,h2] = SCALE * q @ k^T  (batch = 64)
  launch_gemm(Qb, Kb, S, cL, cL, 64, cSCALE, cB * 32,
              (long)cL * 64, (long)cL * 64, (long)cL * cL, stream);

  // ---- differential sign softmax
  sign_softmax<<<cB * 32 * cL, 256, 0, stream>>>(S, mask);
  lambda_kernel<<<1, 32, 0, stream>>>(lq1, lk1, lq2, lk2, lam);
  col_mean<<<dim3(cB * cH, cL / 256), 256, 0, stream>>>(S, G);
  build_att<<<(int)(((long)cB * cH * cL * cL) / 256), 256, 0, stream>>>(S, G, lam, mask, Ab);

  // ---- out = att @ v  (batch = 32)
  launch_gemm(Ab, Vt, O1, cL, 64, cL, 1.f, cB * cH,
              (long)cL * cL, (long)64 * cL, (long)cL * 64, stream);

  // ---- head RMS + relayout, final projection into d_out
  head_rms<<<cB * cH * cL, 64, 0, stream>>>(O1, gh, Ob);
  launch_gemm(Ob, Wot, out, (int)T, cD, cD, 1.f, 1, 0, 0, 0, stream);
}
